// FluidMoELayer_67903432950550
// MI455X (gfx1250) — compile-verified, matrix operand order checked by reading, not run on hardware
//
#include <hip/hip_runtime.h>
#include <math.h>

#define TOKENS  4096
#define HIDDEN  1024
#define FFN_    4096
#define NE      8
#define EF      (NE * FFN_)       // 32768
#define NASSIGN (TOKENS * 2)      // 8192 (token, expert) assignments

typedef __attribute__((ext_vector_type(16))) __bf16   v16bf;
typedef __attribute__((ext_vector_type(8)))  float    v8f;
typedef __attribute__((ext_vector_type(4)))  unsigned v4u;

union Frag { v4u u[2]; v16bf v; };

__device__ __forceinline__ unsigned short f2bf(float f) {
  unsigned u = __float_as_uint(f);
  u += 0x7FFFu + ((u >> 16) & 1u);   // round-to-nearest-even
  return (unsigned short)(u >> 16);
}

// ---------------------------------------------------------------- utilities
__global__ void init_counts_kernel(int* counts) {
  if (threadIdx.x < 16) counts[threadIdx.x] = 0;   // [0..7]=counts, [8..15]=cursors
}

__global__ void conv_f32_bf16_kernel(const float* __restrict__ src,
                                     unsigned short* __restrict__ dst, int n4) {
  int i = blockIdx.x * blockDim.x + threadIdx.x;
  if (i >= n4) return;
  float4 f = ((const float4*)src)[i];
  ushort4 o;
  o.x = f2bf(f.x); o.y = f2bf(f.y); o.z = f2bf(f.z); o.w = f2bf(f.w);
  ((ushort4*)dst)[i] = o;
}

// Fused fp32->bf16 convert + transpose: src [R,C] fp32 -> dst [C,R] bf16.
// 32x32 tiles via LDS; grid (C/32, R/32), block 256.
__global__ void transconv_kernel(const float* __restrict__ src,
                                 unsigned short* __restrict__ dst, int R, int C) {
  __shared__ unsigned short tile[32][33];
  const int c0 = blockIdx.x * 32;
  const int r0 = blockIdx.y * 32;
  const int tc = threadIdx.x & 31;
  const int tr = threadIdx.x >> 5;          // 0..7
#pragma unroll
  for (int i = 0; i < 4; ++i) {
    int r = tr + i * 8;
    tile[r][tc] = f2bf(src[(size_t)(r0 + r) * C + c0 + tc]);
  }
  __syncthreads();
#pragma unroll
  for (int i = 0; i < 4; ++i) {
    int c = tr + i * 8;
    dst[(size_t)(c0 + c) * R + r0 + tc] = tile[tc][c];
  }
}

// ---------------------------------------------------------------- router
__global__ void router_kernel(const float* __restrict__ x, const float* __restrict__ rw,
                              int* __restrict__ counts, int* __restrict__ topk_idx,
                              float* __restrict__ topk_gate) {
  const int t    = blockIdx.x * 4 + (threadIdx.x >> 5);
  const int lane = threadIdx.x & 31;
  float acc[NE];
#pragma unroll
  for (int e = 0; e < NE; ++e) acc[e] = 0.0f;
  const float* xr = x + (size_t)t * HIDDEN;
  for (int h = lane; h < HIDDEN; h += 32) {
    float xv = xr[h];
    const float* r = rw + (size_t)h * NE;
#pragma unroll
    for (int e = 0; e < NE; ++e) acc[e] += xv * r[e];
  }
#pragma unroll
  for (int off = 16; off > 0; off >>= 1)
#pragma unroll
    for (int e = 0; e < NE; ++e) acc[e] += __shfl_xor(acc[e], off, 32);

  if (lane == 0) {
    float m = acc[0];
#pragma unroll
    for (int e = 1; e < NE; ++e) m = fmaxf(m, acc[e]);
    float p[NE], s = 0.0f;
#pragma unroll
    for (int e = 0; e < NE; ++e) { p[e] = expf(acc[e] - m); s += p[e]; }
    float inv = 1.0f / s;
    int i0 = 0; float v0 = p[0];
#pragma unroll
    for (int e = 1; e < NE; ++e) if (p[e] > v0) { v0 = p[e]; i0 = e; }
    int i1 = -1; float v1 = -1.0f;
#pragma unroll
    for (int e = 0; e < NE; ++e) if (e != i0 && p[e] > v1) { v1 = p[e]; i1 = e; }
    topk_idx[t * 2 + 0]  = i0;       topk_idx[t * 2 + 1]  = i1;
    topk_gate[t * 2 + 0] = v0 * inv; topk_gate[t * 2 + 1] = v1 * inv;
    atomicAdd(&counts[i0], 1);
    atomicAdd(&counts[i1], 1);
  }
}

__global__ void scan_kernel(const int* __restrict__ counts, int* __restrict__ offsets) {
  if (threadIdx.x == 0) {
    int acc = 0;
    for (int e = 0; e < NE; ++e) { offsets[e] = acc; acc += counts[e]; }
  }
}

__global__ void fill_kernel(const int* __restrict__ topk_idx,
                            const int* __restrict__ offsets, int* __restrict__ cursors,
                            int* __restrict__ tok_of_slot, int* __restrict__ slot_of) {
  int t = blockIdx.x * blockDim.x + threadIdx.x;
  if (t >= TOKENS) return;
#pragma unroll
  for (int k = 0; k < 2; ++k) {
    int e = topk_idx[t * 2 + k];
    int pos = atomicAdd(&cursors[e], 1);
    int slot = offsets[e] + pos;
    tok_of_slot[slot] = t;
    slot_of[t * 2 + k] = slot;
  }
}

// ---------------------------------------------------------------- grouped GEMMs
// WG tile: M=128 slots x N=64, block 128 (4 waves), wave tile 32x64 (2x4 WMMA).
// Double-buffered LDS, one barrier per K-step of 32.
// A: [row][k] stride 40; B (pre-transposed weights): [n][k] stride 40.

#define LSTR 40

__launch_bounds__(128)
__global__ void ffn1_kernel(const unsigned short* __restrict__ xb,
                            const unsigned short* __restrict__ w1t,   // [E*F, H] bf16
                            const int* __restrict__ counts, const int* __restrict__ offsets,
                            const int* __restrict__ tok_of_slot,
                            unsigned short* __restrict__ hbuf) {
  __shared__ __align__(16) unsigned short Abuf[2][128 * LSTR];
  __shared__ __align__(16) unsigned short Bbuf[2][64 * LSTR];
  __shared__ int tokrow[128];

  int b = blockIdx.x, e = -1, lb = 0;
  {
    int acc = 0;
    for (int i = 0; i < NE; ++i) {
      int nb = (counts[i] + 127) >> 7;
      if (b < acc + nb) { e = i; lb = b - acc; break; }
      acc += nb;
    }
  }
  if (e < 0) return;
  const int cnt  = counts[e];
  const int base = offsets[e];
  const int row0 = lb * 128;
  const int rows = cnt - row0;
  const int tid  = threadIdx.x;

  {
    int r = row0 + tid;
    tokrow[tid] = (r < cnt) ? tok_of_slot[base + r] : 0;
  }
  __syncthreads();

  const int wave = tid >> 5, lane = tid & 31;
  const int half = lane >> 4, r16 = lane & 15;
  const int colblk = blockIdx.y * 64;              // F tile base
  const int ecol = e * FFN_ + colblk;              // row base in w1t

  v8f c0[4] = {}, c1[4] = {};

  // ---- staging helper (unrolled; loads issued before stores) ----
  auto stage = [&](int buf, int k0) {
    v4u ta[4];
    int ra[4], ca[4];
#pragma unroll
    for (int it = 0; it < 4; ++it) {
      int idx = tid + it * 128;                    // 0..511 -> 128 rows x 4 chunks
      ra[it] = idx >> 2; ca[it] = idx & 3;
      ta[it] = *(const v4u*)(xb + (size_t)tokrow[ra[it]] * HIDDEN + k0 + ca[it] * 8);
    }
    v4u tb[2];
    int rb[2], cb[2];
#pragma unroll
    for (int it = 0; it < 2; ++it) {
      int idx = tid + it * 128;                    // 0..255 -> 64 n x 4 chunks
      rb[it] = idx >> 2; cb[it] = idx & 3;
      const unsigned short* src = w1t + (size_t)(ecol + rb[it]) * HIDDEN + k0 + cb[it] * 8;
      tb[it] = *(const v4u*)src;
      if (k0 + 64 < HIDDEN) __builtin_prefetch(src + 64, 0, 0);
    }
#pragma unroll
    for (int it = 0; it < 4; ++it)
      *(v4u*)&Abuf[buf][ra[it] * LSTR + ca[it] * 8] = ta[it];
#pragma unroll
    for (int it = 0; it < 2; ++it)
      *(v4u*)&Bbuf[buf][rb[it] * LSTR + cb[it] * 8] = tb[it];
  };

  stage(0, 0);
  __syncthreads();

  const int NK = HIDDEN / 32;
  const int m0 = wave * 32;
  for (int ks = 0; ks < NK; ++ks) {
    int cur = ks & 1;
    if (ks + 1 < NK) stage(cur ^ 1, (ks + 1) * 32);

    Frag a0, a1;
    const int ar0 = (m0 + r16) * LSTR + half * 8;
    const int ar1 = (m0 + 16 + r16) * LSTR + half * 8;
    a0.u[0] = *(const v4u*)&Abuf[cur][ar0];
    a0.u[1] = *(const v4u*)&Abuf[cur][ar0 + 16];
    a1.u[0] = *(const v4u*)&Abuf[cur][ar1];
    a1.u[1] = *(const v4u*)&Abuf[cur][ar1 + 16];
#pragma unroll
    for (int j = 0; j < 4; ++j) {
      Frag bm;
      const int br = (j * 16 + r16) * LSTR + half * 8;
      bm.u[0] = *(const v4u*)&Bbuf[cur][br];
      bm.u[1] = *(const v4u*)&Bbuf[cur][br + 16];
      c0[j] = __builtin_amdgcn_wmma_f32_16x16x32_bf16(false, a0.v, false, bm.v,
                                                      (short)0, c0[j], false, false);
      c1[j] = __builtin_amdgcn_wmma_f32_16x16x32_bf16(false, a1.v, false, bm.v,
                                                      (short)0, c1[j], false, false);
    }
    __syncthreads();
  }

  // Epilogue: exact GELU, bf16 store. C layout: lane<16 -> M=v, lane>=16 -> M=v+8.
#pragma unroll
  for (int j = 0; j < 4; ++j) {
#pragma unroll
    for (int v = 0; v < 8; ++v) {
      int ncol = colblk + j * 16 + r16;
      int ma = m0 + half * 8 + v;
      if (ma < rows) {
        float xv = c0[j][v];
        float g = 0.5f * xv * (1.0f + erff(xv * 0.70710678118f));
        hbuf[(size_t)(base + row0 + ma) * FFN_ + ncol] = f2bf(g);
      }
      int mb = m0 + 16 + half * 8 + v;
      if (mb < rows) {
        float xv = c1[j][v];
        float g = 0.5f * xv * (1.0f + erff(xv * 0.70710678118f));
        hbuf[(size_t)(base + row0 + mb) * FFN_ + ncol] = f2bf(g);
      }
    }
  }
}

__launch_bounds__(128)
__global__ void ffn2_kernel(const unsigned short* __restrict__ hbuf,
                            const unsigned short* __restrict__ w2t,   // [H, E*F] bf16
                            const int* __restrict__ counts, const int* __restrict__ offsets,
                            float* __restrict__ ybuf) {
  __shared__ __align__(16) unsigned short Abuf[2][128 * LSTR];
  __shared__ __align__(16) unsigned short Bbuf[2][64 * LSTR];

  int b = blockIdx.x, e = -1, lb = 0;
  {
    int acc = 0;
    for (int i = 0; i < NE; ++i) {
      int nb = (counts[i] + 127) >> 7;
      if (b < acc + nb) { e = i; lb = b - acc; break; }
      acc += nb;
    }
  }
  if (e < 0) return;
  const int cnt  = counts[e];
  const int base = offsets[e];
  const int row0 = lb * 128;
  const int rows = cnt - row0;
  const int tid  = threadIdx.x;
  const int wave = tid >> 5, lane = tid & 31;
  const int half = lane >> 4, r16 = lane & 15;
  const int colblk = blockIdx.y * 64;              // H tile base

  v8f c0[4] = {}, c1[4] = {};

  auto stage = [&](int buf, int k0) {
    v4u ta[4];
    int ra[4], ca[4];
#pragma unroll
    for (int it = 0; it < 4; ++it) {
      int idx = tid + it * 128;
      ra[it] = idx >> 2; ca[it] = idx & 3;
      int srow = base + row0 + ra[it]; if (srow > NASSIGN - 1) srow = NASSIGN - 1;
      ta[it] = *(const v4u*)(hbuf + (size_t)srow * FFN_ + k0 + ca[it] * 8);
    }
    v4u tb[2];
    int rb[2], cb[2];
#pragma unroll
    for (int it = 0; it < 2; ++it) {
      int idx = tid + it * 128;
      rb[it] = idx >> 2; cb[it] = idx & 3;
      const unsigned short* src =
          w2t + (size_t)(colblk + rb[it]) * EF + e * FFN_ + k0 + cb[it] * 8;
      tb[it] = *(const v4u*)src;
      if (k0 + 64 < FFN_) __builtin_prefetch(src + 64, 0, 0);
    }
#pragma unroll
    for (int it = 0; it < 4; ++it)
      *(v4u*)&Abuf[buf][ra[it] * LSTR + ca[it] * 8] = ta[it];
#pragma unroll
    for (int it = 0; it < 2; ++it)
      *(v4u*)&Bbuf[buf][rb[it] * LSTR + cb[it] * 8] = tb[it];
  };

  stage(0, 0);
  __syncthreads();

  const int NK = FFN_ / 32;
  const int m0 = wave * 32;
  for (int ks = 0; ks < NK; ++ks) {
    int cur = ks & 1;
    if (ks + 1 < NK) stage(cur ^ 1, (ks + 1) * 32);

    Frag a0, a1;
    const int ar0 = (m0 + r16) * LSTR + half * 8;
    const int ar1 = (m0 + 16 + r16) * LSTR + half * 8;
    a0.u[0] = *(const v4u*)&Abuf[cur][ar0];
    a0.u[1] = *(const v4u*)&Abuf[cur][ar0 + 16];
    a1.u[0] = *(const v4u*)&Abuf[cur][ar1];
    a1.u[1] = *(const v4u*)&Abuf[cur][ar1 + 16];
#pragma unroll
    for (int j = 0; j < 4; ++j) {
      Frag bm;
      const int br = (j * 16 + r16) * LSTR + half * 8;
      bm.u[0] = *(const v4u*)&Bbuf[cur][br];
      bm.u[1] = *(const v4u*)&Bbuf[cur][br + 16];
      c0[j] = __builtin_amdgcn_wmma_f32_16x16x32_bf16(false, a0.v, false, bm.v,
                                                      (short)0, c0[j], false, false);
      c1[j] = __builtin_amdgcn_wmma_f32_16x16x32_bf16(false, a1.v, false, bm.v,
                                                      (short)0, c1[j], false, false);
    }
    __syncthreads();
  }

#pragma unroll
  for (int j = 0; j < 4; ++j) {
#pragma unroll
    for (int v = 0; v < 8; ++v) {
      int ncol = colblk + j * 16 + r16;
      int ma = m0 + half * 8 + v;
      if (ma < rows)
        ybuf[(size_t)(base + row0 + ma) * HIDDEN + ncol] = c0[j][v];
      int mb = m0 + 16 + half * 8 + v;
      if (mb < rows)
        ybuf[(size_t)(base + row0 + mb) * HIDDEN + ncol] = c1[j][v];
    }
  }
}

// ---------------------------------------------------------------- combine (deterministic)
__global__ void combine_kernel(const float* __restrict__ ybuf, const int* __restrict__ slot_of,
                               const float* __restrict__ topk_gate, float* __restrict__ out) {
  int t = blockIdx.x;
  int s0 = slot_of[t * 2], s1 = slot_of[t * 2 + 1];
  float g0 = topk_gate[t * 2], g1 = topk_gate[t * 2 + 1];
  const float* y0 = ybuf + (size_t)s0 * HIDDEN;
  const float* y1 = ybuf + (size_t)s1 * HIDDEN;
  float* o = out + (size_t)t * HIDDEN;
  for (int h = threadIdx.x; h < HIDDEN; h += blockDim.x)
    o[h] = g0 * y0[h] + g1 * y1[h];
}

// ---------------------------------------------------------------- launch
extern "C" void kernel_launch(void* const* d_in, const int* in_sizes, int n_in,
                              void* d_out, int out_size, void* d_ws, size_t ws_size,
                              hipStream_t stream) {
  const float* x  = (const float*)d_in[0];
  const float* rw = (const float*)d_in[1];
  const float* w1 = (const float*)d_in[2];
  const float* w2 = (const float*)d_in[3];
  (void)in_sizes; (void)n_in; (void)out_size; (void)ws_size;

  char* p = (char*)d_ws;
  auto alloc = [&](size_t bytes) -> char* {
    char* r = p; p += (bytes + 255) & ~(size_t)255; return r;
  };
  unsigned short* xb   = (unsigned short*)alloc((size_t)TOKENS * HIDDEN * 2); // 8 MB
  unsigned short* w1t  = (unsigned short*)alloc((size_t)HIDDEN * EF * 2);     // 64 MB  [EF, H]
  unsigned short* w2t  = (unsigned short*)alloc((size_t)EF * HIDDEN * 2);     // 64 MB  [H, EF]
  unsigned short* hbuf = (unsigned short*)alloc((size_t)NASSIGN * FFN_ * 2);  // 64 MB
  float*          ybuf = (float*)alloc((size_t)NASSIGN * HIDDEN * 4);         // 32 MB
  int*   counts    = (int*)alloc(16 * sizeof(int));    // [0..7]=counts, [8..15]=cursors
  int*   offsets   = (int*)alloc(8 * sizeof(int));
  int*   topk_idx  = (int*)alloc((size_t)TOKENS * 2 * sizeof(int));
  float* topk_gate = (float*)alloc((size_t)TOKENS * 2 * sizeof(float));
  int*   tok_of_slot = (int*)alloc((size_t)NASSIGN * sizeof(int));
  int*   slot_of     = (int*)alloc((size_t)TOKENS * 2 * sizeof(int));

  init_counts_kernel<<<1, 32, 0, stream>>>(counts);

  int n4x = TOKENS * HIDDEN / 4;
  conv_f32_bf16_kernel<<<(n4x + 255) / 256, 256, 0, stream>>>(x, xb, n4x);
  // w1 [H, EF] -> w1t [EF, H];  w2 [EF, H] -> w2t [H, EF]
  transconv_kernel<<<dim3(EF / 32, HIDDEN / 32), 256, 0, stream>>>(w1, w1t, HIDDEN, EF);
  transconv_kernel<<<dim3(HIDDEN / 32, EF / 32), 256, 0, stream>>>(w2, w2t, EF, HIDDEN);

  router_kernel<<<TOKENS / 4, 128, 0, stream>>>(x, rw, counts, topk_idx, topk_gate);
  scan_kernel<<<1, 1, 0, stream>>>(counts, offsets);
  fill_kernel<<<TOKENS / 256, 256, 0, stream>>>(topk_idx, offsets, counts + 8,
                                                tok_of_slot, slot_of);

  // sum_e ceil(cnt_e/128) <= 8192/128 + 8 = 72 blocks in M
  ffn1_kernel<<<dim3(72, FFN_ / 64), 128, 0, stream>>>(xb, w1t, counts, offsets,
                                                       tok_of_slot, hbuf);
  ffn2_kernel<<<dim3(72, HIDDEN / 64), 128, 0, stream>>>(hbuf, w2t, counts, offsets, ybuf);

  combine_kernel<<<TOKENS, 256, 0, stream>>>(ybuf, slot_of, topk_gate, (float*)d_out);
}
